// MixtralDecoderLayer_26379689132541
// MI455X (gfx1250) — matrix operand order repaired and checked by reading, not compiled, not run on hardware
//
#include <hip/hip_runtime.h>
#include <math.h>

// ---------------- problem constants ----------------
#define H    1024
#define NH   16
#define NKV  4
#define HD   64
#define NE   8
#define FF   2048
#define TT   2048
#define QKVN ((NH + 2 * NKV) * HD)   // 1536
#define EPSV 1e-5f

typedef _Float16 h16;
typedef __attribute__((ext_vector_type(16))) _Float16 v16h;
typedef __attribute__((ext_vector_type(8)))  _Float16 v8h;
typedef __attribute__((ext_vector_type(8)))  float    v8f;

// ---------------- WMMA helpers ----------------
// A fragment (16x32 f16, MxK): lane l holds row m=(l&15); halves:
// elems 0..7  = K[kb .. kb+7],  elems 8..15 = K[kb+8 .. kb+15], kb=(l>>4)*16
__device__ inline v16h load_a_frag(const h16* base, int ld) {
  int l = threadIdx.x & 31;
  const h16* row = base + (size_t)(l & 15) * ld + ((l >> 4) * 16);
  v8h lo = *(const v8h*)(row);
  v8h hi = *(const v8h*)(row + 8);
  v16h a;
#pragma unroll
  for (int i = 0; i < 8; ++i) { a[i] = lo[i]; a[8 + i] = hi[i]; }
  return a;
}

// B fragment (32x16 f16, KxN), loaded from B^T stored row-major [N][K]:
// lane l holds column n=(l&15), k = (l>>4)*16 + 0..15 contiguous.
__device__ inline v16h load_b_frag(const h16* baseT, int ld) {
  int l = threadIdx.x & 31;
  const h16* p = baseT + (size_t)(l & 15) * ld + ((l >> 4) * 16);
  return *(const v16h*)p;
}

__device__ inline v8f wmma_f16(v16h a, v16h b, v8f c) {
  return __builtin_amdgcn_wmma_f32_16x16x32_f16(false, a, false, b,
                                                (short)0, c, false, false);
}

// ---------------- CDNA5 async memory->LDS copy (ASYNCcnt path) ----------------
// Per-lane 16B DMA into LDS, no VGPR data round-trip; tracked by ASYNCcnt.
__device__ inline void async_copy_b128(const h16* g, unsigned lds_off) {
  asm volatile("global_load_async_to_lds_b128 %0, %1, off"
               :
               : "v"(lds_off), "v"((unsigned long long)g)
               : "memory");
}
__device__ inline void wait_async_le3() {
  asm volatile("s_wait_asynccnt 0x3" ::: "memory");
}
__device__ inline void wait_async_0() {
  asm volatile("s_wait_asynccnt 0x0" ::: "memory");
}
__device__ inline unsigned lds_offset(const void* p) {
  // generic pointer to LDS carries the LDS byte offset in addr[31:0]
  return (unsigned)(size_t)p;
}

// ---------------- f32 -> f16 convert ----------------
__global__ __launch_bounds__(256) void cvt_f32_f16_kernel(
    const float* __restrict__ s, h16* __restrict__ d, long long n) {
  long long i = ((long long)blockIdx.x * 256 + threadIdx.x) * 8;
  if (i >= n) return;
  v8h o;
#pragma unroll
  for (int j = 0; j < 8; ++j) o[j] = (h16)s[i + j];
  *(v8h*)(d + i) = o;
}

// ---------------- rmsnorm ----------------
__global__ __launch_bounds__(256) void rmsnorm_kernel(
    const float* __restrict__ x, const float* __restrict__ w,
    float* __restrict__ of, h16* __restrict__ oh) {
  __shared__ float red[256];
  int t = blockIdx.x;
  const float* row = x + (size_t)t * H;
  float s = 0.f;
  for (int i = threadIdx.x; i < H; i += 256) { float v = row[i]; s += v * v; }
  red[threadIdx.x] = s;
  __syncthreads();
  for (int off = 128; off > 0; off >>= 1) {
    if (threadIdx.x < off) red[threadIdx.x] += red[threadIdx.x + off];
    __syncthreads();
  }
  float rstd = rsqrtf(red[0] / (float)H + EPSV);
  for (int i = threadIdx.x; i < H; i += 256) {
    float v = row[i] * rstd * w[i];
    if (of) of[(size_t)t * H + i] = v;
    if (oh) oh[(size_t)t * H + i] = (h16)v;
  }
}

// ---- WMMA GEMM: C[M,N] = A[M,K] * W[N,K]^T, double-buffered async staging ----
#define BM 128
#define BN 64
#define BK 32
#define ABUF (BM * BK * 2)   // bytes per A buffer (8 KB)
#define BBUF (BN * BK * 2)   // bytes per B buffer (4 KB)

__global__ __launch_bounds__(256) void gemm_f16_kernel(
    const h16* __restrict__ A, const h16* __restrict__ W,
    float* __restrict__ C, int M, int N, int K) {
  __shared__ h16 As[2][BM * BK];   // 2 x 8 KB
  __shared__ h16 Bs[2][BN * BK];   // 2 x 4 KB
  int tid = threadIdx.x;
  int wid = tid >> 5;
  int mbase = blockIdx.y * BM;
  int nbase = blockIdx.x * BN;
  int wm = (wid >> 1) * 32;        // 4 waves along M
  int wn = (wid & 1) * 32;         // 2 waves along N

  // per-thread staging assignment: rows ra and ra+64 of A tile, row ra of B tile
  int ra = tid >> 2;               // 0..63
  int ka = (tid & 3) * 8;          // 0,8,16,24 (f16 elems) -> 16B chunks
  const h16* pa0 = A + (size_t)(mbase + ra) * K + ka;
  const h16* pa1 = pa0 + (size_t)64 * K;
  const h16* pb  = W + (size_t)(nbase + ra) * K + ka;
  unsigned la = lds_offset(&As[0][ra * BK + ka]);
  unsigned lb = lds_offset(&Bs[0][ra * BK + ka]);

  v8f acc[2][2] = {};

  // prime the pipeline: tile 0 -> buffer 0
  async_copy_b128(pa0, la);
  async_copy_b128(pa1, la + 64 * BK * 2);
  async_copy_b128(pb,  lb);
  pa0 += BK; pa1 += BK; pb += BK;

  int buf = 0;
  for (int k0 = 0; k0 < K; k0 += BK) {
    bool more = (k0 + BK) < K;
    if (more) {  // overlap next tile's DMA with this tile's WMMAs
      unsigned nb = (unsigned)(buf ^ 1);
      async_copy_b128(pa0, la + nb * ABUF);
      async_copy_b128(pa1, la + nb * ABUF + 64 * BK * 2);
      async_copy_b128(pb,  lb + nb * BBUF);
      pa0 += BK; pa1 += BK; pb += BK;
      wait_async_le3();   // previous tile's 3 async ops complete (in-order)
    } else {
      wait_async_0();
    }
    __syncthreads();
    const h16* Ab = &As[buf][0];
    const h16* Bb = &Bs[buf][0];
#pragma unroll
    for (int sm = 0; sm < 2; ++sm) {
      v16h af = load_a_frag(Ab + (wm + sm * 16) * BK, BK);
#pragma unroll
      for (int sn = 0; sn < 2; ++sn) {
        v16h bf = load_b_frag(Bb + (wn + sn * 16) * BK, BK);
        acc[sm][sn] = wmma_f16(af, bf, acc[sm][sn]);
      }
    }
    __syncthreads();   // all waves done with buf before it is refilled
    buf ^= 1;
  }

  int l = tid & 31;
#pragma unroll
  for (int sm = 0; sm < 2; ++sm)
#pragma unroll
    for (int sn = 0; sn < 2; ++sn)
#pragma unroll
      for (int i = 0; i < 8; ++i) {
        int r  = mbase + wm + sm * 16 + (l >> 4) * 8 + i;
        int cc = nbase + wn + sn * 16 + (l & 15);
        C[(size_t)r * N + cc] = acc[sm][sn][i];
      }
}

// ---------------- RoPE on q/k + V transpose ----------------
__global__ __launch_bounds__(256) void qkv_post_kernel(
    const float* __restrict__ qkv, const int* __restrict__ pos,
    h16* __restrict__ Qh, h16* __restrict__ Kh, h16* __restrict__ VTh) {
  int t = blockIdx.x;
  float p = (float)pos[t];
  const float* row = qkv + (size_t)t * QKVN;
  for (int idx = threadIdx.x; idx < NH * 32; idx += 256) {
    int hh = idx >> 5, j = idx & 31;
    float inv = __powf(10000.0f, -(float)j / 32.0f);
    float a = p * inv, c = __cosf(a), sn = __sinf(a);
    float x1 = row[hh * HD + j], x2 = row[hh * HD + 32 + j];
    h16* q = Qh + ((size_t)hh * TT + t) * HD;
    q[j]      = (h16)(x1 * c - x2 * sn);
    q[32 + j] = (h16)(x2 * c + x1 * sn);
  }
  for (int idx = threadIdx.x; idx < NKV * 32; idx += 256) {
    int kv = idx >> 5, j = idx & 31;
    float inv = __powf(10000.0f, -(float)j / 32.0f);
    float a = p * inv, c = __cosf(a), sn = __sinf(a);
    float x1 = row[NH * HD + kv * HD + j];
    float x2 = row[NH * HD + kv * HD + 32 + j];
    h16* kp = Kh + ((size_t)kv * TT + t) * HD;
    kp[j]      = (h16)(x1 * c - x2 * sn);
    kp[32 + j] = (h16)(x2 * c + x1 * sn);
  }
  for (int idx = threadIdx.x; idx < NKV * HD; idx += 256) {
    int kv = idx >> 6, d = idx & 63;
    VTh[((size_t)kv * HD + d) * TT + t] =
        (h16)row[(NH + NKV) * HD + kv * HD + d];
  }
}

// ---------------- causal flash attention, 1 wave per (head, 16 queries) ----
__global__ __launch_bounds__(32) void attn_kernel(
    const h16* __restrict__ Qh, const h16* __restrict__ Kh,
    const h16* __restrict__ VTh, h16* __restrict__ O) {
  __shared__ h16 Ps[16 * 32];
  int l = threadIdx.x & 31;
  int qb = blockIdx.x, h = blockIdx.y;
  int kvh = h >> 2;                     // NH/NKV = 4
  int qbase = qb * 16;
  const h16* Q  = Qh  + ((size_t)h * TT + qbase) * HD;
  const h16* Kp = Kh  + (size_t)kvh * TT * HD;   // [T][64]
  const h16* Vt = VTh + (size_t)kvh * HD * TT;   // [64][T]

  v16h qa0 = load_a_frag(Q, HD);        // dims 0..31
  v16h qa1 = load_a_frag(Q + 32, HD);   // dims 32..63

  v8f o0 = {}, o1 = {}, o2 = {}, o3 = {};
  float mrow[8], lrow[8];
#pragma unroll
  for (int i = 0; i < 8; ++i) { mrow[i] = -1e30f; lrow[i] = 0.f; }

  int n = l & 15, half = l >> 4;
  int nkb = ((qbase + 15) >> 5) + 1;
  for (int kb = 0; kb < nkb; ++kb) {
    int kbb = kb * 32;
    v8f s0 = {}, s1 = {};
    s0 = wmma_f16(qa0, load_b_frag(Kp + (size_t)kbb * HD, HD), s0);
    s0 = wmma_f16(qa1, load_b_frag(Kp + (size_t)kbb * HD + 32, HD), s0);
    s1 = wmma_f16(qa0, load_b_frag(Kp + (size_t)(kbb + 16) * HD, HD), s1);
    s1 = wmma_f16(qa1, load_b_frag(Kp + (size_t)(kbb + 16) * HD + 32, HD), s1);
#pragma unroll
    for (int i = 0; i < 8; ++i) {       // scale + causal mask
      int r = qbase + half * 8 + i;
      float v0 = s0[i] * 0.125f;        // HD^-0.5
      float v1 = s1[i] * 0.125f;
      if (kbb + n > r)      v0 = -1e30f;
      if (kbb + 16 + n > r) v1 = -1e30f;
      s0[i] = v0; s1[i] = v1;
    }
#pragma unroll
    for (int i = 0; i < 8; ++i) {       // online softmax per row
      float mx = fmaxf(s0[i], s1[i]);
#pragma unroll
      for (int d = 8; d >= 1; d >>= 1) mx = fmaxf(mx, __shfl_xor(mx, d));
      float mnew = fmaxf(mrow[i], mx);
      float p0 = __expf(s0[i] - mnew);
      float p1 = __expf(s1[i] - mnew);
      float rs = p0 + p1;
#pragma unroll
      for (int d = 8; d >= 1; d >>= 1) rs += __shfl_xor(rs, d);
      float corr = __expf(mrow[i] - mnew);
      lrow[i] = lrow[i] * corr + rs;
      mrow[i] = mnew;
      o0[i] *= corr; o1[i] *= corr; o2[i] *= corr; o3[i] *= corr;
      Ps[(half * 8 + i) * 32 + n]      = (h16)p0;   // re-layout via LDS
      Ps[(half * 8 + i) * 32 + 16 + n] = (h16)p1;
    }
    v16h pa = load_a_frag(Ps, 32);      // per-wave in-order DS: RAW safe
    o0 = wmma_f16(pa, load_b_frag(Vt + (size_t)0 * 16 * TT + kbb, TT), o0);
    o1 = wmma_f16(pa, load_b_frag(Vt + (size_t)1 * 16 * TT + kbb, TT), o1);
    o2 = wmma_f16(pa, load_b_frag(Vt + (size_t)2 * 16 * TT + kbb, TT), o2);
    o3 = wmma_f16(pa, load_b_frag(Vt + (size_t)3 * 16 * TT + kbb, TT), o3);
  }
#pragma unroll
  for (int i = 0; i < 8; ++i) {
    float inv = 1.0f / lrow[i];
    int r = qbase + half * 8 + i;
    h16* orow = O + (size_t)r * (NH * HD) + h * HD;
    orow[ 0 + n] = (h16)(o0[i] * inv);
    orow[16 + n] = (h16)(o1[i] * inv);
    orow[32 + n] = (h16)(o2[i] * inv);
    orow[48 + n] = (h16)(o3[i] * inv);
  }
}

// ---------------- elementwise ----------------
__global__ __launch_bounds__(256) void add_kernel(
    const float* __restrict__ a, const float* __restrict__ b,
    float* __restrict__ c, long long n) {
  long long i = (long long)blockIdx.x * 256 + threadIdx.x;
  if (i < n) c[i] = a[i] + b[i];
}

__global__ __launch_bounds__(256) void silu_mul_kernel(
    const float* __restrict__ h1, const float* __restrict__ h3,
    h16* __restrict__ act, long long n) {
  long long i = (long long)blockIdx.x * 256 + threadIdx.x;
  if (i >= n) return;
  float a = h1[i];
  float s = a / (1.f + __expf(-a));
  act[i] = (h16)(s * h3[i]);
}

__global__ __launch_bounds__(256) void moe_accum_kernel(
    const float* __restrict__ y, const float* __restrict__ wmoe,
    float* __restrict__ acc, int e, int init) {
  long long i = (long long)blockIdx.x * 256 + threadIdx.x;
  if (i >= (long long)TT * H) return;
  long long t = i / H;
  float v = wmoe[t * NE + e] * y[i];
  acc[i] = init ? v : acc[i] + v;
}

// ---------------- MoE gate: softmax + top-2 -> dense weights ----------------
__global__ __launch_bounds__(256) void gate_kernel(
    const float* __restrict__ xn, const float* __restrict__ gw,
    float* __restrict__ wmoe) {
  __shared__ float lg[NE];
  int t = blockIdx.x;
  int wv = threadIdx.x >> 5, l = threadIdx.x & 31;
  const float* row = xn + (size_t)t * H;
  const float* g = gw + (size_t)wv * H;
  float s = 0.f;
  for (int i = l; i < H; i += 32) s += row[i] * g[i];
#pragma unroll
  for (int d = 16; d >= 1; d >>= 1) s += __shfl_xor(s, d);
  if (l == 0) lg[wv] = s;
  __syncthreads();
  if (threadIdx.x == 0) {
    float mx = lg[0];
    for (int e = 1; e < NE; ++e) mx = fmaxf(mx, lg[e]);
    float pe[NE], sum = 0.f;
    for (int e = 0; e < NE; ++e) { pe[e] = __expf(lg[e] - mx); sum += pe[e]; }
    for (int e = 0; e < NE; ++e) pe[e] /= sum;
    int a0 = 0;
    for (int e = 1; e < NE; ++e) if (pe[e] > pe[a0]) a0 = e;
    int a1 = (a0 == 0) ? 1 : 0;
    for (int e = 0; e < NE; ++e) if (e != a0 && pe[e] > pe[a1]) a1 = e;
    float ws = pe[a0] + pe[a1];
    for (int e = 0; e < NE; ++e)
      wmoe[(size_t)t * NE + e] =
          (e == a0) ? pe[a0] / ws : (e == a1 ? pe[a1] / ws : 0.f);
  }
}

// ---------------- host orchestration ----------------
extern "C" void kernel_launch(void* const* d_in, const int* in_sizes, int n_in,
                              void* d_out, int out_size, void* d_ws,
                              size_t ws_size, hipStream_t stream) {
  const int*   pos       = (const int*)d_in[0];
  const float* hidden    = (const float*)d_in[1];
  const float* w_qkv     = (const float*)d_in[2];
  const float* w_o       = (const float*)d_in[3];
  const float* norm_in   = (const float*)d_in[4];
  const float* norm_post = (const float*)d_in[5];
  const float* norm_next = (const float*)d_in[6];
  const float* gate_w    = (const float*)d_in[7];
  const float* w1        = (const float*)d_in[8];
  const float* w2        = (const float*)d_in[9];
  const float* w3        = (const float*)d_in[10];
  float* out  = (float*)d_out;
  float* xout = out + (size_t)TT * H;   // second tuple element

  char* ws = (char*)d_ws;
  size_t off = 0;
  auto alloc = [&](size_t bytes) -> void* {
    void* p = ws + off;
    off = (off + bytes + 255) & ~(size_t)255;
    return p;
  };

  h16* wqkv_h = (h16*)alloc((size_t)QKVN * H * 2);
  h16* wo_h   = (h16*)alloc((size_t)H * (NH * HD) * 2);
  h16* w1_h   = (h16*)alloc((size_t)NE * FF * H * 2);
  h16* w3_h   = (h16*)alloc((size_t)NE * FF * H * 2);
  h16* w2_h   = (h16*)alloc((size_t)NE * H * FF * 2);
  h16* xn1_h  = (h16*)alloc((size_t)TT * H * 2);
  float* qkv_f = (float*)alloc((size_t)TT * QKVN * 4);
  h16* q_h    = (h16*)alloc((size_t)NH * TT * HD * 2);
  h16* k_h    = (h16*)alloc((size_t)NKV * TT * HD * 2);
  h16* vT_h   = (h16*)alloc((size_t)NKV * HD * TT * 2);
  h16* attn_h = (h16*)alloc((size_t)TT * NH * HD * 2);
  float* aproj_f = (float*)alloc((size_t)TT * H * 4);
  float* x1_f    = (float*)alloc((size_t)TT * H * 4);
  h16* xn2_h     = (h16*)alloc((size_t)TT * H * 2);
  float* xn2_f   = (float*)alloc((size_t)TT * H * 4);
  float* wmoe_f  = (float*)alloc((size_t)TT * NE * 4);
  float* h1_f    = (float*)alloc((size_t)TT * FF * 4);
  float* h3_f    = (float*)alloc((size_t)TT * FF * 4);
  h16* act_h     = (h16*)alloc((size_t)TT * FF * 2);
  float* y_f     = (float*)alloc((size_t)TT * H * 4);
  float* moe_f   = (float*)alloc((size_t)TT * H * 4);
  (void)ws_size; (void)in_sizes; (void)n_in; (void)out_size;

  auto cvt = [&](const float* s, h16* d, long long n) {
    cvt_f32_f16_kernel<<<(unsigned)((n / 8 + 255) / 256), 256, 0, stream>>>(s, d, n);
  };
  auto gemm = [&](const h16* A, const h16* W, float* C, int M, int N, int K) {
    gemm_f16_kernel<<<dim3(N / BN, M / BM), 256, 0, stream>>>(A, W, C, M, N, K);
  };

  // weight conversion (f32 -> f16)
  cvt(w_qkv, wqkv_h, (long long)QKVN * H);
  cvt(w_o,   wo_h,   (long long)H * NH * HD);
  cvt(w1,    w1_h,   (long long)NE * FF * H);
  cvt(w3,    w3_h,   (long long)NE * FF * H);
  cvt(w2,    w2_h,   (long long)NE * H * FF);

  // attention block
  rmsnorm_kernel<<<TT, 256, 0, stream>>>(hidden, norm_in, nullptr, xn1_h);
  gemm(xn1_h, wqkv_h, qkv_f, TT, QKVN, H);
  qkv_post_kernel<<<TT, 256, 0, stream>>>(qkv_f, pos, q_h, k_h, vT_h);
  attn_kernel<<<dim3(TT / 16, NH), 32, 0, stream>>>(q_h, k_h, vT_h, attn_h);
  gemm(attn_h, wo_h, aproj_f, TT, H, NH * HD);
  add_kernel<<<(TT * H) / 256, 256, 0, stream>>>(hidden, aproj_f, x1_f,
                                                 (long long)TT * H);

  // MoE block (dense over experts, matching reference contraction)
  rmsnorm_kernel<<<TT, 256, 0, stream>>>(x1_f, norm_post, xn2_f, xn2_h);
  gate_kernel<<<TT, 256, 0, stream>>>(xn2_f, gate_w, wmoe_f);
  for (int e = 0; e < NE; ++e) {
    gemm(xn2_h, w1_h + (size_t)e * FF * H, h1_f, TT, FF, H);
    gemm(xn2_h, w3_h + (size_t)e * FF * H, h3_f, TT, FF, H);
    silu_mul_kernel<<<(TT * FF) / 256, 256, 0, stream>>>(h1_f, h3_f, act_h,
                                                         (long long)TT * FF);
    gemm(act_h, w2_h + (size_t)e * H * FF, y_f, TT, H, FF);
    moe_accum_kernel<<<(TT * H) / 256, 256, 0, stream>>>(y_f, wmoe_f, moe_f, e,
                                                         e == 0 ? 1 : 0);
  }
  add_kernel<<<(TT * H) / 256, 256, 0, stream>>>(x1_f, moe_f, xout,
                                                 (long long)TT * H);
  rmsnorm_kernel<<<TT, 256, 0, stream>>>(xout, norm_next, out, nullptr);
}
